// Archer_2001454760747
// MI455X (gfx1250) — matrix operand order repaired and checked by reading, not compiled, not run on hardware
//
#include <hip/hip_runtime.h>

// ---------------------------------------------------------------------------
// CDNA5 (gfx1250) implementation of the ST-GNN forward pass.
// All dense contractions run on v_wmma_f32_16x16x32_f16 (wave32 WMMA),
// attention is fused (scores+softmax+PV in LDS) to avoid the 0.25GB score
// tensor, everything else is bandwidth-trivial elementwise work.
// ---------------------------------------------------------------------------

typedef _Float16 f16;
typedef _Float16 v16h __attribute__((ext_vector_type(16)));
typedef _Float16 v8h  __attribute__((ext_vector_type(8)));
typedef float    v8f  __attribute__((ext_vector_type(8)));

#define BN_SCALE 0.9999950000374997f  /* 1/sqrt(1+1e-5) */
#define NP 416                         /* 400 padded to 13*32 */

__device__ __forceinline__ v16h jn(v8h a, v8h b) {
  v16h r;
#pragma unroll
  for (int i = 0; i < 8; ++i) { r[i] = a[i]; r[i + 8] = b[i]; }
  return r;
}

__device__ __forceinline__ v8f wmma32(v16h a, v16h b, v8f c) {
  return __builtin_amdgcn_wmma_f32_16x16x32_f16(false, a, false, b, (short)0, c,
                                                false, false);
}

// ---------------------------------------------------------------------------
// Generic batched WMMA GEMM:
//   C[z][r][c] = act( sum_k A(z,r,k) * Bt[z][c*ldBt + k] + bias[c] )
// A(z,r,k) = A[z*aBat + rowbase(r) + (k>>6)*tap + (k&63)]
// rowbase(r): r -> (b,t,n) over (rows = B*Tout*Nn), base=((b*Tin+t)*Nn+n)*ldA.
// Plain GEMM: Tin=Tout=1, Nn=rows, tap=64  ->  base = r*ldA, k identity.
// Temporal conv (kernel k): Kdim=k*64, tap=Nn*64 walks taps through time.
// ---------------------------------------------------------------------------
__global__ __launch_bounds__(256)
void gemm_k(const f16* __restrict__ A, const f16* __restrict__ Bt,
            const float* __restrict__ bias,
            float* __restrict__ Cf, f16* __restrict__ Ch,
            int ldC, int rows, int Kdim, int ncols,
            int ldA, int ldBt, int tap,
            int Tin, int Tout, int Nn, int act,
            long aBat, int aDiv, long bBat, int bDiv, long cBat)
{
  const int lane = threadIdx.x & 31;
  const int rowTiles = rows >> 4;
  const int tiles = rowTiles * (ncols >> 4);
  const int wid = blockIdx.x * (blockDim.x >> 5) + (threadIdx.x >> 5);
  if (wid >= tiles) return;                       // uniform per wave
  const int rt = wid % rowTiles;
  const int ct = wid / rowTiles;
  const int z  = blockIdx.y;
  A  += (long)(z / aDiv) * aBat;
  Bt += (long)(z / bDiv) * bBat;
  const long coff = (long)z * cBat;

  const int row0 = rt << 4, col0 = ct << 4;
  const int bI = row0 / (Tout * Nn);
  const int tI = (row0 / Nn) % Tout;
  const int n0 = row0 % Nn;
  const long abase = ((long)(bI * Tin + tI) * Nn + n0 + (lane & 15)) * ldA;
  const f16* brow = Bt + (long)(col0 + (lane & 15)) * ldBt;

  const int o8  = (lane >= 16) ? 8 : 0;
  const int o16 = (lane >= 16) ? 16 : 0;
  v8f acc = {0.f, 0.f, 0.f, 0.f, 0.f, 0.f, 0.f, 0.f};

  for (int k0 = 0; k0 < Kdim; k0 += 32) {
    const int ka = k0 + o8;
    const int kb = ka + 16;
    v8h a0 = *(const v8h*)(A + abase + (long)(ka >> 6) * tap + (ka & 63));
    v8h a1 = *(const v8h*)(A + abase + (long)(kb >> 6) * tap + (kb & 63));
    const int kc = k0 + o8;
    v8h b0 = *(const v8h*)(brow + kc);
    v8h b1 = *(const v8h*)(brow + kc + 16);
    if (k0 + 32 < Kdim)
      __builtin_prefetch(A + abase + (long)((k0 + 32) >> 6) * tap + ((k0 + 32) & 63), 0, 0);
    acc = wmma32(jn(a0, a1), jn(b0, b1), acc);
  }

  const float bvv = bias ? bias[col0 + (lane & 15)] : 0.f;
#pragma unroll
  for (int r = 0; r < 8; ++r) {
    const int m = r + o8;
    float v = acc[r] + bvv;
    if (act == 1) v = v > 0.f ? v : 0.f;
    const long o = coff + (long)(row0 + m) * ldC + col0 + (lane & 15);
    if (Cf) Cf[o] = v;
    if (Ch) Ch[o] = (f16)v;
  }
}

// ---------------------------------------------------------------------------
// Fused spatial attention, one block per (b,t,h). 128 threads = 4 waves.
// scores S = K Q^T / 20 (WMMA, frags straight from global QKV),
// softmax rows in LDS, O = P V (WMMA, P from LDS).
// QKV layout: row (b*T+t)*400 + n, 768 cols = [Q(0..255) K(256..511) V(512..767)].
// ---------------------------------------------------------------------------
__global__ __launch_bounds__(128)
void attn_k(const f16* __restrict__ qkv, f16* __restrict__ oout, float scale)
{
  __shared__ __align__(16) float S[16 * NP];
  __shared__ __align__(16) f16   P[16 * NP];
  __shared__ float red[128];
  __shared__ float rmax[16], rsum[16];

  const int tid = threadIdx.x, lane = tid & 31, w = tid >> 5;
  const int z = blockIdx.x;          // (b*T+t)*H + h
  const int h = z & 3;
  const long bt = z >> 2;
  const f16* Qp = qkv + bt * 400 * 768 +   0 + h * 64;
  const f16* Kp = qkv + bt * 400 * 768 + 256 + h * 64;
  const f16* Vp = qkv + bt * 400 * 768 + 512 + h * 64;

  for (int i = tid; i < 16 * 16; i += 128)       // zero P pads (cols 400..415)
    P[(i >> 4) * NP + 400 + (i & 15)] = (f16)0.f;
  __syncthreads();

  const int o8  = (lane >= 16) ? 8 : 0;
  const int o16 = (lane >= 16) ? 16 : 0;

  for (int it = 0; it < 25; ++it) {
    const int i0 = it * 16;
    // ---- scores: wave w handles j-tiles w, w+4, ... ----
    for (int jt = w; jt < 25; jt += 4) {
      const int j0 = jt * 16;
      const f16* arow = Kp + (long)(i0 + (lane & 15)) * 768;
      const f16* brow = Qp + (long)(j0 + (lane & 15)) * 768;
      v8f acc = {0.f, 0.f, 0.f, 0.f, 0.f, 0.f, 0.f, 0.f};
#pragma unroll
      for (int k0 = 0; k0 < 64; k0 += 32) {
        v8h a0 = *(const v8h*)(arow + k0 + o8);
        v8h a1 = *(const v8h*)(arow + k0 + o8 + 16);
        v8h b0 = *(const v8h*)(brow + k0 + o8);
        v8h b1 = *(const v8h*)(brow + k0 + o8 + 16);
        acc = wmma32(jn(a0, a1), jn(b0, b1), acc);
      }
#pragma unroll
      for (int r = 0; r < 8; ++r)
        S[(r + o8) * NP + j0 + (lane & 15)] = acc[r] * scale;
    }
    __syncthreads();

    // ---- softmax: 8 threads per row ----
    const int row = tid & 15, sub = tid >> 4;
    float mx = -1e30f;
    for (int j = sub; j < 400; j += 8) mx = fmaxf(mx, S[row * NP + j]);
    red[tid] = mx;
    __syncthreads();
    if (sub == 0) {
      float m2 = red[row];
      for (int s = 1; s < 8; ++s) m2 = fmaxf(m2, red[row + (s << 4)]);
      rmax[row] = m2;
    }
    __syncthreads();
    mx = rmax[row];
    float sum = 0.f;
    for (int j = sub; j < 400; j += 8) {
      float e = __expf(S[row * NP + j] - mx);
      S[row * NP + j] = e;
      sum += e;
    }
    red[tid] = sum;
    __syncthreads();
    if (sub == 0) {
      float s2 = 0.f;
      for (int s = 0; s < 8; ++s) s2 += red[row + (s << 4)];
      rsum[row] = s2;
    }
    __syncthreads();
    const float inv = 1.f / rsum[row];
    for (int j = sub; j < 400; j += 8)
      P[row * NP + j] = (f16)(S[row * NP + j] * inv);
    __syncthreads();

    // ---- O = P @ V: wave w computes output cols m0..m0+15 ----
    {
      const int m0 = w * 16;
      v8f acc = {0.f, 0.f, 0.f, 0.f, 0.f, 0.f, 0.f, 0.f};
      for (int k0 = 0; k0 < NP; k0 += 32) {
        v8h a0 = *(const v8h*)(&P[(lane & 15) * NP + k0 + o8]);
        v8h a1 = *(const v8h*)(&P[(lane & 15) * NP + k0 + o8 + 16]);
        v16h bfr;
#pragma unroll
        for (int e = 0; e < 16; ++e) {
          int j = k0 + o8 + (e < 8 ? e : e + 8); j = j < 400 ? j : 399;   // pad: P==0 anyway
          bfr[e] = Vp[(long)j * 768 + m0 + (lane & 15)];
        }
        acc = wmma32(jn(a0, a1), bfr, acc);
      }
#pragma unroll
      for (int r = 0; r < 8; ++r)
        oout[(bt * 400 + i0 + r + o8) * 256 + h * 64 + m0 + (lane & 15)] =
            (f16)acc[r];
    }
    __syncthreads();
  }
}

// ----------------------------- elementwise kernels -------------------------
__global__ void incr_k(const float* __restrict__ x, const float* __restrict__ w,
                       const float* __restrict__ b, float* __restrict__ of,
                       f16* __restrict__ oh, long n)
{
  long i = (long)blockIdx.x * blockDim.x + threadIdx.x;
  if (i >= n) return;
  long r = i >> 6; int c = (int)(i & 63);
  float v = x[r * 2] * w[c] + x[r * 2 + 1] * w[64 + c] + b[c];
  of[i] = v; oh[i] = (f16)v;
}

__global__ void tcnc_k(const float* __restrict__ C1, const float* __restrict__ C2,
                       const float* __restrict__ g, const float* __restrict__ bt,
                       const f16* __restrict__ rA, int rATin, int rAoff,
                       const f16* __restrict__ rB, int rBTin, int rBoff,
                       f16* __restrict__ ho, int Tout, long n)
{
  long i = (long)blockIdx.x * blockDim.x + threadIdx.x;
  if (i >= n) return;
  int c = (int)(i & 63); long r = i >> 6;
  int nn = (int)(r % 400); long rt = r / 400;
  int t = (int)(rt % Tout); int b = (int)(rt / Tout);
  float a = C1[i]; a = a > 0.f ? a : 0.f;
  float s = 1.f / (1.f + __expf(-C2[i]));
  float v = a * s * (g[c] * (float)BN_SCALE) + bt[c];
  if (rA) v += (float)rA[(((long)(b * rATin + t + rAoff) * 400) + nn) * 64 + c];
  if (rB) v += (float)rB[(((long)(b * rBTin + t + rBoff) * 400) + nn) * 64 + c];
  ho[i] = (f16)v;
}

__global__ void ln_reduce_k(const float* __restrict__ C, const float* __restrict__ resid,
                            float* __restrict__ stats, int perB)
{
  __shared__ float ss[256], sq[256];
  const int b = blockIdx.x; const long base = (long)b * perB;
  float s = 0.f, q = 0.f;
  for (int i = threadIdx.x; i < perB; i += 256) {
    float v = C[base + i] + resid[base + i];
    s += v; q += v * v;
  }
  ss[threadIdx.x] = s; sq[threadIdx.x] = q; __syncthreads();
  for (int o = 128; o > 0; o >>= 1) {
    if (threadIdx.x < o) { ss[threadIdx.x] += ss[threadIdx.x + o];
                           sq[threadIdx.x] += sq[threadIdx.x + o]; }
    __syncthreads();
  }
  if (threadIdx.x == 0) {
    float mu = ss[0] / perB;
    stats[b * 2] = mu; stats[b * 2 + 1] = sq[0] / perB - mu * mu;
  }
}

__global__ void ln_apply_k(const float* __restrict__ C, const float* __restrict__ resid,
                           const float* __restrict__ stats, const float* __restrict__ g,
                           const float* __restrict__ bta, float* __restrict__ o,
                           int perB, long n)
{
  long i = (long)blockIdx.x * blockDim.x + threadIdx.x;
  if (i >= n) return;
  int b = (int)(i / perB); int j = (int)(i % perB);
  float rs = rsqrtf(stats[b * 2 + 1] + 1e-5f);
  o[i] = (C[i] + resid[i] - stats[b * 2]) * rs * g[j] + bta[j];
}

__global__ void hg_k(const float* __restrict__ on, const float* __restrict__ wtr,
                     const float* __restrict__ btr, f16* __restrict__ hg, long n)
{
  long i = (long)blockIdx.x * blockDim.x + threadIdx.x;
  if (i >= n) return;
  int m = (int)(i & 63); long r = i >> 6;
  int nn = (int)(r % 400); int b = (int)(r / 400);
  float v = btr[0];
#pragma unroll
  for (int t = 0; t < 12; ++t)
    v += on[(((long)(b * 12 + t) * 400) + nn) * 64 + m] * wtr[t];
  hg[i] = (f16)(v > 0.f ? v : 0.f);
}

__global__ void resft_k(const float* __restrict__ e, const float* __restrict__ wtc,
                        const float* __restrict__ btc, f16* __restrict__ out, long n)
{ // layout out[b][p][m][j] with j padded to 416
  long i = (long)blockIdx.x * blockDim.x + threadIdx.x;
  if (i >= n) return;
  int j = (int)(i % NP); long r = i / NP;
  int m = (int)(r % 64); long r2 = r / 64;
  int p = (int)(r2 % 12); int b = (int)(r2 / 12);
  float v = 0.f;
  if (j < 400) {
    v = e[((long)b * 400 + j) * 64 + m] * wtc[p] + btc[p];
    v = v > 0.f ? v : 0.f;
  }
  out[i] = (f16)v;
}

__global__ void smat_k(const f16* __restrict__ adjh, const float* __restrict__ AA,
                       f16* __restrict__ sh, long n)
{ // n = 8*400*400
  long i = (long)blockIdx.x * blockDim.x + threadIdx.x;
  if (i >= n) return;
  int j = (int)(i % 400); long r = i / 400;
  long o = r * NP + j;
  sh[o] = (f16)((float)adjh[o] + 2.f * AA[o]);
}

__global__ void pred_k(const f16* __restrict__ g2, const float* __restrict__ wp,
                       const float* __restrict__ bp, float* __restrict__ out, long n)
{
  long r = (long)blockIdx.x * blockDim.x + threadIdx.x;
  if (r >= n) return;
  float v = bp[0];
#pragma unroll
  for (int m = 0; m < 64; ++m) v += (float)g2[r * 64 + m] * wp[m];
  out[r] = v;
}

__global__ void wt_k(const float* __restrict__ W, f16* __restrict__ Bt, int K, int N)
{ // Bt[c*K+k] = W[k*N+c]
  long i = (long)blockIdx.x * blockDim.x + threadIdx.x;
  if (i >= (long)K * N) return;
  int k = (int)(i % K); long c = i / K;
  Bt[i] = (f16)W[(long)k * N + c];
}

__global__ void fillh_k(f16* __restrict__ p, long n)
{
  long i = (long)blockIdx.x * blockDim.x + threadIdx.x;
  if (i < n) p[i] = (f16)0.f;
}

// ---------------------------------------------------------------------------
extern "C" void kernel_launch(void* const* d_in, const int* in_sizes, int n_in,
                              void* d_out, int out_size, void* d_ws, size_t ws_size,
                              hipStream_t stream)
{
  const float* x      = (const float*)d_in[0];
  const float* w_inc  = (const float*)d_in[1];
  const float* b_inc  = (const float*)d_in[2];
  const float* Wq     = (const float*)d_in[3];
  const float* bq     = (const float*)d_in[4];
  const float* Wk     = (const float*)d_in[5];
  const float* bk     = (const float*)d_in[6];
  const float* Wv     = (const float*)d_in[7];
  const float* bv     = (const float*)d_in[8];
  const float* Wo     = (const float*)d_in[9];
  const float* bo     = (const float*)d_in[10];
  const float* ln_g   = (const float*)d_in[11];
  const float* ln_b   = (const float*)d_in[12];
  const float* w_tr   = (const float*)d_in[13];
  const float* b_tr   = (const float*)d_in[14];
  const float* gcn_w  = (const float*)d_in[15];
  const float* gcn_b  = (const float*)d_in[16];
  const float* t3w1   = (const float*)d_in[17];
  const float* t3b1   = (const float*)d_in[18];
  const float* t3w2   = (const float*)d_in[19];
  const float* t3b2   = (const float*)d_in[20];
  const float* t3g    = (const float*)d_in[21];
  const float* t3bt   = (const float*)d_in[22];
  const float* t2w1   = (const float*)d_in[23];
  const float* t2b1   = (const float*)d_in[24];
  const float* t2w2   = (const float*)d_in[25];
  const float* t2b2   = (const float*)d_in[26];
  const float* t2g    = (const float*)d_in[27];
  const float* t2bt   = (const float*)d_in[28];
  const float* W_full = (const float*)d_in[29];
  const float* b_full = (const float*)d_in[30];
  const float* w_tc   = (const float*)d_in[31];
  const float* b_tc   = (const float*)d_in[32];
  const float* W_lin  = (const float*)d_in[33];
  const float* b_lin  = (const float*)d_in[34];
  const float* W_pred = (const float*)d_in[35];
  const float* b_pred = (const float*)d_in[36];
  float* out = (float*)d_out;

  const long RT = 38400;  // B*T*N

  // ---- workspace carve ----
  char* wsb = (char*)d_ws; size_t off = 0;
  auto alloc = [&](size_t bytes) -> void* {
    void* p = wsb + off;
    off = (off + bytes + 255) & ~(size_t)255;
    return p;
  };
  float* out_f  = (float*)alloc(RT * 64 * 4);
  f16*   out_h  = (f16*)  alloc(RT * 64 * 2);
  f16*   qkv    = (f16*)  alloc(RT * 768 * 2);
  f16*   attn_o = (f16*)  alloc(RT * 256 * 2);
  float* woC    = (float*)alloc(RT * 64 * 4);
  float* onorm  = (float*)alloc(RT * 64 * 4);
  float* stats  = (float*)alloc(64);
  f16*   hg_h   = (f16*)  alloc((long)8 * 400 * 64 * 2);
  f16*   adj_h  = (f16*)  alloc((long)8 * 400 * NP * 2);
  float* AAb    = (float*)alloc((long)8 * 400 * NP * 4);
  f16*   smat_h = (f16*)  alloc((long)8 * 400 * NP * 2);
  float* e_f    = (float*)alloc((long)8 * 400 * 64 * 4);
  f16*   resft  = (f16*)  alloc((long)8 * 12 * 64 * NP * 2);
  f16*   lfs_h  = (f16*)  alloc(RT * 64 * 2);
  f16*   g_h    = (f16*)  alloc(RT * 64 * 2);
  f16*   g2_h   = (f16*)  alloc(RT * 64 * 2);
  float* C1     = (float*)alloc((long)32000 * 64 * 4);
  float* C2     = (float*)alloc((long)32000 * 64 * 4);
  f16*   hA     = (f16*)  alloc((long)32000 * 64 * 2);
  f16*   hB     = (f16*)  alloc((long)32000 * 64 * 2);
  f16*   wqT    = (f16*)  alloc((long)256 * 64 * 2);
  f16*   wkT    = (f16*)  alloc((long)256 * 64 * 2);
  f16*   wvT    = (f16*)  alloc((long)256 * 64 * 2);
  f16*   woT    = (f16*)  alloc((long)64 * 256 * 2);
  f16*   w1T    = (f16*)  alloc((long)5 * 64 * 192 * 2);
  f16*   w2T    = (f16*)  alloc((long)5 * 64 * 192 * 2);
  f16*   t2w1T  = (f16*)  alloc((long)64 * 128 * 2);
  f16*   t2w2T  = (f16*)  alloc((long)64 * 128 * 2);
  f16*   wfullT = (f16*)  alloc((long)64 * 64 * 2);
  f16*   gcnwT  = (f16*)  alloc((long)8 * 64 * 64 * 2);
  f16*   wlinT  = (f16*)  alloc((long)64 * 64 * 2);
  (void)ws_size; (void)in_sizes; (void)n_in; (void)out_size;

  auto G1 = [](long n) { return dim3((unsigned)((n + 255) / 256)); };
  auto gemm = [&](const f16* Ap, const f16* Btp, const float* bp,
                  float* Cf, f16* Ch, int ldC, int rows, int K, int ncols,
                  int ldA, int ldBt, int tap, int Tin, int Tout, int Nn, int act,
                  int nbat, long aBat, int aDiv, long bBat, int bDiv, long cBat) {
    int tiles = (rows / 16) * (ncols / 16);
    dim3 grid((unsigned)((tiles + 7) / 8), (unsigned)nbat);
    gemm_k<<<grid, 256, 0, stream>>>(Ap, Btp, bp, Cf, Ch, ldC, rows, K, ncols,
                                     ldA, ldBt, tap, Tin, Tout, Nn, act,
                                     aBat, aDiv, bBat, bDiv, cBat);
  };

  // ---- weight prep (f32 -> f16 transposed Bt[c][k]) ----
  wt_k<<<G1(64 * 256), 256, 0, stream>>>(Wq, wqT, 64, 256);
  wt_k<<<G1(64 * 256), 256, 0, stream>>>(Wk, wkT, 64, 256);
  wt_k<<<G1(64 * 256), 256, 0, stream>>>(Wv, wvT, 64, 256);
  wt_k<<<G1(256 * 64), 256, 0, stream>>>(Wo, woT, 256, 64);
  for (int i = 0; i < 5; ++i) {
    wt_k<<<G1(192 * 64), 256, 0, stream>>>(t3w1 + (long)i * 192 * 64, w1T + (long)i * 64 * 192, 192, 64);
    wt_k<<<G1(192 * 64), 256, 0, stream>>>(t3w2 + (long)i * 192 * 64, w2T + (long)i * 64 * 192, 192, 64);
  }
  wt_k<<<G1(128 * 64), 256, 0, stream>>>(t2w1, t2w1T, 128, 64);
  wt_k<<<G1(128 * 64), 256, 0, stream>>>(t2w2, t2w2T, 128, 64);
  wt_k<<<G1(64 * 64), 256, 0, stream>>>(W_full, wfullT, 64, 64);
  for (int b = 0; b < 8; ++b)
    wt_k<<<G1(64 * 64), 256, 0, stream>>>(gcn_w + b * 4096, gcnwT + b * 4096, 64, 64);
  wt_k<<<G1(64 * 64), 256, 0, stream>>>(W_lin, wlinT, 64, 64);

  // zero padded adjacency buffers (K=416 contractions read the pads)
  fillh_k<<<G1((long)8 * 400 * NP), 256, 0, stream>>>(adj_h, (long)8 * 400 * NP);
  fillh_k<<<G1((long)8 * 400 * NP), 256, 0, stream>>>(smat_h, (long)8 * 400 * NP);

  // ---- incrDimention: out = x @ w_inc + b_inc ----
  incr_k<<<G1(RT * 64), 256, 0, stream>>>(x, w_inc, b_inc, out_f, out_h, RT * 64);

  // ---- QKV projections into one [row][768] buffer ----
  gemm(out_h, wqT, bq, nullptr, qkv + 0,   768, (int)RT, 64, 256, 64, 64, 64, 1, 1, (int)RT, 0, 1, 0, 1, 0, 1, 0);
  gemm(out_h, wkT, bk, nullptr, qkv + 256, 768, (int)RT, 64, 256, 64, 64, 64, 1, 1, (int)RT, 0, 1, 0, 1, 0, 1, 0);
  gemm(out_h, wvT, bv, nullptr, qkv + 512, 768, (int)RT, 64, 256, 64, 64, 64, 1, 1, (int)RT, 0, 1, 0, 1, 0, 1, 0);

  // ---- fused attention, one block per (b,t,h) ----
  attn_k<<<dim3(384), dim3(128), 0, stream>>>(qkv, attn_o, 0.05f);

  // ---- o = attn_o @ Wo + bo (+ residual & LayerNorm) ----
  gemm(attn_o, woT, bo, woC, nullptr, 64, (int)RT, 256, 64, 256, 256, 64, 1, 1, (int)RT, 0, 1, 0, 1, 0, 1, 0);
  ln_reduce_k<<<dim3(8), 256, 0, stream>>>(woC, out_f, stats, 307200);
  ln_apply_k<<<G1(RT * 64), 256, 0, stream>>>(woC, out_f, stats, ln_g, ln_b, onorm, 307200, RT * 64);

  // ---- TCN encoder (on out_h) ----
  const f16* cur = out_h; int Tin = 12; f16* hbufs[2] = {hA, hB}; int pb = 0;
  for (int i = 0; i < 5; ++i) {
    int Tout = Tin - 2; long rows = (long)8 * Tout * 400;
    gemm(cur, w1T + (long)i * 64 * 192, t3b1 + i * 64, C1, nullptr, 64, (int)rows,
         192, 64, 64, 192, 400 * 64, Tin, Tout, 400, 0, 1, 0, 1, 0, 1, 0);
    gemm(cur, w2T + (long)i * 64 * 192, t3b2 + i * 64, C2, nullptr, 64, (int)rows,
         192, 64, 64, 192, 400 * 64, Tin, Tout, 400, 0, 1, 0, 1, 0, 1, 0);
    const f16* rA = (i == 0) ? out_h : cur;
    int rATin = (i == 0) ? 12 : Tin;
    const f16* rB = (i == 0) ? nullptr : out_h;
    tcnc_k<<<G1(rows * 64), 256, 0, stream>>>(C1, C2, t3g + i * 64, t3bt + i * 64,
        rA, rATin, 2, rB, 12, 2 * (i + 1), hbufs[pb], Tout, rows * 64);
    cur = hbufs[pb]; pb ^= 1; Tin = Tout;
  }
  { // final kernel-2 layer: Tin=2 -> Tout=1
    long rows = 3200;
    gemm(cur, t2w1T, t2b1, C1, nullptr, 64, (int)rows, 128, 64, 64, 128, 400 * 64, 2, 1, 400, 0, 1, 0, 1, 0, 1, 0);
    gemm(cur, t2w2T, t2b2, C2, nullptr, 64, (int)rows, 128, 64, 64, 128, 400 * 64, 2, 1, 400, 0, 1, 0, 1, 0, 1, 0);
    tcnc_k<<<G1(rows * 64), 256, 0, stream>>>(C1, C2, t2g, t2bt,
        cur, 2, 1, out_h, 12, 11, hbufs[pb], 1, rows * 64);
    cur = hbufs[pb];
  }
  // e = relu(h0 @ W_full + b_full)
  gemm(cur, wfullT, b_full, e_f, nullptr, 64, 3200, 64, 64, 64, 64, 64, 1, 1, 3200, 1, 1, 0, 1, 0, 1, 0);
  // resf transposed: [b][p][m][j(416)]
  resft_k<<<G1((long)8 * 12 * 64 * NP), 256, 0, stream>>>(e_f, w_tc, b_tc, resft, (long)8 * 12 * 64 * NP);

  // ---- graph branch ----
  hg_k<<<G1((long)8 * 400 * 64), 256, 0, stream>>>(onorm, w_tr, b_tr, hg_h, (long)8 * 400 * 64);
  // adj = hg hg^T (symmetric) -> adj_h padded f16
  gemm(hg_h, hg_h, nullptr, nullptr, adj_h, NP, 400, 64, 400, 64, 64, 64, 1, 1, 400, 0,
       8, (long)400 * 64, 1, (long)400 * 64, 1, (long)400 * NP);
  // AA = adj @ adj (adj symmetric -> Bt = adj)
  gemm(adj_h, adj_h, nullptr, AAb, nullptr, NP, 400, NP, 400, NP, NP, 64, 1, 1, 400, 0,
       8, (long)400 * NP, 1, (long)400 * NP, 1, (long)400 * NP);
  smat_k<<<G1((long)8 * 400 * 400), 256, 0, stream>>>(adj_h, AAb, smat_h, (long)8 * 400 * 400);
  // lfs[b][p] = S[b] @ resf[b][p]   (z = b*12+p, A batch = z/12)
  gemm(smat_h, resft, nullptr, nullptr, lfs_h, 64, 400, NP, 64, NP, NP, 64, 1, 1, 400, 0,
       96, (long)400 * NP, 12, (long)64 * NP, 1, (long)400 * 64);
  // g = relu(lfs @ gcn_w[b] + gcn_b)
  gemm(lfs_h, gcnwT, gcn_b, nullptr, g_h, 64, 4800, 64, 64, 64, 64, 64, 1, 1, 4800, 1,
       8, (long)4800 * 64, 1, (long)64 * 64, 1, (long)4800 * 64);
  // g2 = relu(g @ W_lin + b_lin)
  gemm(g_h, wlinT, b_lin, nullptr, g2_h, 64, (int)RT, 64, 64, 64, 64, 64, 1, 1, (int)RT, 1, 1, 0, 1, 0, 1, 0);
  // prediction head (64 -> 1)
  pred_k<<<G1(RT), 256, 0, stream>>>(g2_h, W_pred, b_pred, out, RT);
}
